// ReprogrammingLayerV2_80169859547972
// MI455X (gfx1250) — matrix operand order repaired and checked by reading, not compiled, not run on hardware
//
#include <hip/hip_runtime.h>
#include <hip/hip_bf16.h>

// ---------------------------------------------------------------------------
// ReprogrammingLayerV2 for MI455X (gfx1250): LN -> QKV proj -> flash attn ->
// mean-pool -> out proj.  All GEMMs on v_wmma_f32_16x16x32_bf16 (wave32).
// Softmax row reductions use DPP16 row_ror rotations (VALU); the bank is
// streamed in 64-wide chunks so the per-chunk softmax bookkeeping is
// amortized over 16 WMMAs per iteration.
// ---------------------------------------------------------------------------

typedef __bf16 bf16;
typedef __attribute__((ext_vector_type(16))) __bf16 v16bf;
typedef __attribute__((ext_vector_type(8)))  __bf16 v8bf;
typedef __attribute__((ext_vector_type(8)))  float  v8f;

#define D_   768
#define H_   12
#define E_   64
#define S_   2000
#define SP_  2048              // S padded to multiple of 64 (GEMM/attn tiles)
#define B_   4
#define L_   64
#define TQ_  8
#define NQ_  (B_ * L_ * TQ_)   // 2048 query rows
#define NBL_ (B_ * L_)         // 256 pooled rows

static __device__ __forceinline__ v8f wmma_bf16(v16bf a, v16bf b, v8f c) {
  // D = A(16x32 bf16) x B(32x16 bf16) + C(16x16 f32)
  return __builtin_amdgcn_wmma_f32_16x16x32_bf16(
      /*neg_a=*/false, a, /*neg_b=*/false, b,
      /*c_mod=*/(short)0, c, /*reuse_a=*/false, /*reuse_b=*/false);
}

// ---- DPP16 row rotation (within each 16-lane row of the wave) -------------
// ctrl = 0x120 | N  ->  row_ror:N   (gfx10+ DPP16 encoding family)
template <int CTRL>
static __device__ __forceinline__ float dpp_rot(float x) {
  int r = __builtin_amdgcn_update_dpp(0, __float_as_int(x), CTRL,
                                      /*row_mask=*/0xf, /*bank_mask=*/0xf,
                                      /*bound_ctrl=*/true);
  return __int_as_float(r);
}
static __device__ __forceinline__ float row_max16(float t) {
  t = fmaxf(t, dpp_rot<0x121>(t));   // ror:1
  t = fmaxf(t, dpp_rot<0x122>(t));   // ror:2
  t = fmaxf(t, dpp_rot<0x124>(t));   // ror:4
  t = fmaxf(t, dpp_rot<0x128>(t));   // ror:8
  return t;
}
static __device__ __forceinline__ float row_sum16(float t) {
  t += dpp_rot<0x121>(t);
  t += dpp_rot<0x122>(t);
  t += dpp_rot<0x124>(t);
  t += dpp_rot<0x128>(t);
  return t;
}

// ---------------------------------------------------------------------------
// LayerNorm -> bf16.  One 256-thread block per row (D=768 = 3 per thread).
// Rows >= valid_rows are written as zeros (padding for the bank).
// ---------------------------------------------------------------------------
__global__ __launch_bounds__(256)
void k_ln(const float* __restrict__ x, const float* __restrict__ g,
          const float* __restrict__ beta, bf16* __restrict__ out,
          int valid_rows) {
  const int row = blockIdx.x;
  const int tid = threadIdx.x;
  bf16* orow = out + (size_t)row * D_;
  if (row >= valid_rows) {
    for (int d = tid; d < D_; d += 256) orow[d] = (bf16)0.0f;
    return;
  }
  const float* xr = x + (size_t)row * D_;
  float lx[3];
  float s = 0.0f;
#pragma unroll
  for (int j = 0; j < 3; ++j) { lx[j] = xr[tid + j * 256]; s += lx[j]; }

  __shared__ float red[256];
  red[tid] = s;
  __syncthreads();
  for (int off = 128; off > 0; off >>= 1) {
    if (tid < off) red[tid] += red[tid + off];
    __syncthreads();
  }
  const float mean = red[0] * (1.0f / D_);
  __syncthreads();

  float sq = 0.0f;
#pragma unroll
  for (int j = 0; j < 3; ++j) { float t = lx[j] - mean; sq += t * t; }
  red[tid] = sq;
  __syncthreads();
  for (int off = 128; off > 0; off >>= 1) {
    if (tid < off) red[tid] += red[tid + off];
    __syncthreads();
  }
  const float rstd = rsqrtf(red[0] * (1.0f / D_) + 1e-5f);

#pragma unroll
  for (int j = 0; j < 3; ++j) {
    const int d = tid + j * 256;
    orow[d] = (bf16)((lx[j] - mean) * rstd * g[d] + beta[d]);
  }
}

// ---------------------------------------------------------------------------
// f32 -> bf16 weight conversion.
// ---------------------------------------------------------------------------
__global__ __launch_bounds__(256)
void k_cvt(const float* __restrict__ in, bf16* __restrict__ out, int n) {
  const int i = blockIdx.x * 256 + threadIdx.x;
  if (i < n) out[i] = (bf16)in[i];
}

// ---------------------------------------------------------------------------
// C(MxN) = A(MxK) * W(NxK)^T + bias.  bf16 inputs, f32 accumulate.
// Block = 128 threads (4 waves); 64x64 output tile; wave w owns rows w*16..+15.
// OUT_MODE: 0 = bf16 row-major, 1 = bf16 transposed (C[n*ldc+m], for V^T),
//           2 = f32 row-major (final projection -> d_out).
// ---------------------------------------------------------------------------
template <int OUT_MODE>
__global__ __launch_bounds__(128)
void k_gemm(const bf16* __restrict__ A, const bf16* __restrict__ W,
            const float* __restrict__ bias, bf16* __restrict__ Cb,
            float* __restrict__ Cf, int M, int N, int K, int ldc) {
  const int lane = threadIdx.x & 31;
  const int wave = threadIdx.x >> 5;
  const int half = lane >> 4;   // which 16-lane group
  const int nn   = lane & 15;
  const int m0 = blockIdx.x * 64 + wave * 16;
  const int n0 = blockIdx.y * 64;

  v8f acc[4] = {v8f{}, v8f{}, v8f{}, v8f{}};
  const bf16* Arow = A + (size_t)(m0 + nn) * K;   // lane's A row (M = nn)

  for (int k0 = 0; k0 < K; k0 += 32) {
    // A fragment: lane-half picks K offset 0/16; elements cover
    // K = half*16 + e (contiguous, matching B).
    v8bf alo = *(const v8bf*)(Arow + k0 + half * 16);
    v8bf ahi = *(const v8bf*)(Arow + k0 + half * 16 + 8);
    v16bf a;
#pragma unroll
    for (int e = 0; e < 8; ++e) { a[e] = alo[e]; a[e + 8] = ahi[e]; }

#pragma unroll
    for (int jn = 0; jn < 4; ++jn) {
      // B fragment: lane = column N, elements = K = half*16 + e (contiguous).
      v16bf b = *(const v16bf*)(W + (size_t)(n0 + jn * 16 + nn) * K +
                                k0 + half * 16);
      acc[jn] = wmma_bf16(a, b, acc[jn]);
    }
  }

#pragma unroll
  for (int jn = 0; jn < 4; ++jn) {
    const int col = n0 + jn * 16 + nn;
    const float bv = bias[col];
#pragma unroll
    for (int i = 0; i < 8; ++i) {
      const int row = m0 + i + half * 8;       // C layout: VGPR i, lane half
      const float v = acc[jn][i] + bv;
      if (OUT_MODE == 0)      Cb[(size_t)row * ldc + col] = (bf16)v;
      else if (OUT_MODE == 1) Cb[(size_t)col * ldc + row] = (bf16)v;
      else                    Cf[(size_t)row * ldc + col] = v;
    }
  }
}

// ---------------------------------------------------------------------------
// Flash attention: one wave per (16-query tile, head).  Streams the bank in
// 64-wide chunks: 8 score WMMAs, one online-softmax pass (DPP16 row
// reductions, one oacc rescale), P (16x64) staged in LDS as bf16, then
// 8 PV WMMAs against the transposed V.  Writes un-pooled O (NQ x D, f32).
// ---------------------------------------------------------------------------
__global__ __launch_bounds__(32)
void k_attn(const bf16* __restrict__ Q, const bf16* __restrict__ Kp,
            const bf16* __restrict__ Vt, const int* __restrict__ mask,
            float* __restrict__ O) {
  const int lane = threadIdx.x & 31;
  const int half = lane >> 4;
  const int nn   = lane & 15;
  const int q0    = blockIdx.x * 16;
  const int hbase = blockIdx.y * E_;
  const float scale = 0.125f;          // 1/sqrt(E), E = 64
  const float NEG  = -1.0e30f;

  __shared__ __align__(32) bf16 ldsP[16 * 64];

  // Q A-fragments for K-dim chunks c=0 (dims 0..31) and c=1 (dims 32..63).
  const bf16* qrow = Q + (size_t)(q0 + nn) * D_ + hbase;
  v16bf a0, a1;
  {
    v8bf l0 = *(const v8bf*)(qrow + half * 16);
    v8bf h0 = *(const v8bf*)(qrow + half * 16 + 8);
    v8bf l1 = *(const v8bf*)(qrow + 32 + half * 16);
    v8bf h1 = *(const v8bf*)(qrow + 40 + half * 16);
#pragma unroll
    for (int e = 0; e < 8; ++e) {
      a0[e] = l0[e]; a0[e + 8] = h0[e];
      a1[e] = l1[e]; a1[e + 8] = h1[e];
    }
  }

  float mrun[8], lrun[8];
#pragma unroll
  for (int i = 0; i < 8; ++i) { mrun[i] = NEG; lrun[i] = 0.0f; }
  v8f oacc[4] = {v8f{}, v8f{}, v8f{}, v8f{}};

  for (int s0 = 0; s0 < SP_; s0 += 64) {
    // prefetch next chunk's K/V stream (global_prefetch_b8)
    if (s0 + 64 < SP_) {
#pragma unroll
      for (int j = 0; j < 4; ++j)
        __builtin_prefetch(Kp + (size_t)(s0 + 64 + j * 16 + nn) * D_ + hbase,
                           0, 1);
      __builtin_prefetch(Vt + (size_t)(hbase + nn) * SP_ + s0 + 64, 0, 1);
    }

    // ---- scores: Q(16x64) x K_chunk^T -> four 16x16 tiles ----------------
    v8f sc[4];
#pragma unroll
    for (int j = 0; j < 4; ++j) {
      const bf16* krow = Kp + (size_t)(s0 + j * 16 + nn) * D_ + hbase;
      v16bf b0 = *(const v16bf*)(krow + half * 16);
      v16bf b1 = *(const v16bf*)(krow + 32 + half * 16);
      v8f c = v8f{};
      c = wmma_bf16(a0, b0, c);
      c = wmma_bf16(a1, b1, c);
      sc[j] = c;
    }

    // per-column mask / scale
    float msc[4], madd[4];
#pragma unroll
    for (int j = 0; j < 4; ++j) {
      const int col = s0 + j * 16 + nn;
      const bool valid = (col < S_) && (mask[col] != 0);
      msc[j]  = valid ? scale : 0.0f;
      madd[j] = valid ? 0.0f : NEG;
    }

    // ---- online softmax: one reduction set per row per 64 columns --------
#pragma unroll
    for (int i = 0; i < 8; ++i) {
      float x[4];
      float loc = NEG;
#pragma unroll
      for (int j = 0; j < 4; ++j) {
        x[j] = sc[j][i] * msc[j] + madd[j];
        loc = fmaxf(loc, x[j]);
      }
      const float t = row_max16(loc);
      const float mnew  = fmaxf(mrun[i], t);
      const float alpha = __expf(mrun[i] - mnew);
      float ls = 0.0f;
      const int prow = i + half * 8;                 // C-layout row
#pragma unroll
      for (int j = 0; j < 4; ++j) {
        const float p = __expf(x[j] - mnew);
        ls += p;
        ldsP[prow * 64 + j * 16 + nn] = (bf16)p;
      }
      const float rs = row_sum16(ls);
      lrun[i] = lrun[i] * alpha + rs;
      mrun[i] = mnew;
#pragma unroll
      for (int jn = 0; jn < 4; ++jn) oacc[jn][i] *= alpha;
    }
    // same-wave LDS RAW: drain DS stores before re-reading as A-fragments
    asm volatile("s_wait_dscnt 0x0" ::: "memory");

    // ---- P (16x64) as two A-fragments from LDS ---------------------------
    v16bf pa0, pa1;
    {
      const bf16* pr = ldsP + nn * 64;
      v8bf l0 = *(const v8bf*)(pr + half * 16);
      v8bf h0 = *(const v8bf*)(pr + half * 16 + 8);
      v8bf l1 = *(const v8bf*)(pr + 32 + half * 16);
      v8bf h1 = *(const v8bf*)(pr + 40 + half * 16);
#pragma unroll
      for (int e = 0; e < 8; ++e) {
        pa0[e] = l0[e]; pa0[e + 8] = h0[e];
        pa1[e] = l1[e]; pa1[e + 8] = h1[e];
      }
    }

    // ---- O += P x V_chunk (V stored transposed: row = dim, col = s) ------
#pragma unroll
    for (int jn = 0; jn < 4; ++jn) {
      const bf16* vrow = Vt + (size_t)(hbase + jn * 16 + nn) * SP_ + s0;
      v16bf vb0 = *(const v16bf*)(vrow + half * 16);
      v16bf vb1 = *(const v16bf*)(vrow + 32 + half * 16);
      oacc[jn] = wmma_bf16(pa0, vb0, oacc[jn]);
      oacc[jn] = wmma_bf16(pa1, vb1, oacc[jn]);
    }
  }

  // normalize and store f32 O
#pragma unroll
  for (int i = 0; i < 8; ++i) {
    const float inv = 1.0f / lrun[i];
    const int row = q0 + i + half * 8;
#pragma unroll
    for (int jn = 0; jn < 4; ++jn)
      O[(size_t)row * D_ + hbase + jn * 16 + nn] = oacc[jn][i] * inv;
  }
}

// ---------------------------------------------------------------------------
// Mean-pool over TQ -> bf16 (input to the output projection GEMM).
// ---------------------------------------------------------------------------
__global__ __launch_bounds__(256)
void k_pool(const float* __restrict__ O, bf16* __restrict__ P) {
  const int idx = blockIdx.x * 256 + threadIdx.x;
  if (idx >= NBL_ * D_) return;
  const int bl = idx / D_, d = idx % D_;
  float s = 0.0f;
#pragma unroll
  for (int t = 0; t < TQ_; ++t) s += O[(size_t)(bl * TQ_ + t) * D_ + d];
  P[idx] = (bf16)(s * (1.0f / TQ_));
}

// ---------------------------------------------------------------------------
extern "C" void kernel_launch(void* const* d_in, const int* in_sizes, int n_in,
                              void* d_out, int out_size, void* d_ws,
                              size_t ws_size, hipStream_t stream) {
  const float* tgt   = (const float*)d_in[0];
  const float* kbank = (const float*)d_in[1];
  const float* vbank = (const float*)d_in[2];
  const int*   bmask = (const int*)d_in[3];
  const float* Wq = (const float*)d_in[4];
  const float* bq = (const float*)d_in[5];
  const float* Wk = (const float*)d_in[6];
  const float* bk = (const float*)d_in[7];
  const float* Wv = (const float*)d_in[8];
  const float* bv = (const float*)d_in[9];
  const float* Wo = (const float*)d_in[10];
  const float* bo = (const float*)d_in[11];
  const float* g_q  = (const float*)d_in[12];
  const float* be_q = (const float*)d_in[13];
  const float* g_kv  = (const float*)d_in[14];
  const float* be_kv = (const float*)d_in[15];

  char* ws = (char*)d_ws;
  size_t off = 0;
  auto carve = [&](size_t bytes) -> char* {
    char* p = ws + off;
    off += (bytes + 255) & ~(size_t)255;
    return p;
  };
  bf16* Qln = (bf16*)carve((size_t)NQ_ * D_ * 2);
  bf16* Kln = (bf16*)carve((size_t)SP_ * D_ * 2);
  bf16* Vln = (bf16*)carve((size_t)SP_ * D_ * 2);
  bf16* Wqb = (bf16*)carve((size_t)D_ * D_ * 2);
  bf16* Wkb = (bf16*)carve((size_t)D_ * D_ * 2);
  bf16* Wvb = (bf16*)carve((size_t)D_ * D_ * 2);
  bf16* Wob = (bf16*)carve((size_t)D_ * D_ * 2);
  bf16* Qp  = (bf16*)carve((size_t)NQ_ * D_ * 2);
  bf16* Kp  = (bf16*)carve((size_t)SP_ * D_ * 2);
  bf16* Vt  = (bf16*)carve((size_t)D_ * SP_ * 2);   // transposed: [dim][s]
  float* O  = (float*)carve((size_t)NQ_ * D_ * 4);
  bf16* Pl  = (bf16*)carve((size_t)NBL_ * D_ * 2);

  // 1) LayerNorms (bank padded to SP_ rows, pads zeroed)
  k_ln<<<NQ_, 256, 0, stream>>>(tgt, g_q, be_q, Qln, NQ_);
  k_ln<<<SP_, 256, 0, stream>>>(kbank, g_kv, be_kv, Kln, S_);
  k_ln<<<SP_, 256, 0, stream>>>(vbank, g_kv, be_kv, Vln, S_);

  // 2) weights -> bf16
  const int NW = D_ * D_;
  k_cvt<<<(NW + 255) / 256, 256, 0, stream>>>(Wq, Wqb, NW);
  k_cvt<<<(NW + 255) / 256, 256, 0, stream>>>(Wk, Wkb, NW);
  k_cvt<<<(NW + 255) / 256, 256, 0, stream>>>(Wv, Wvb, NW);
  k_cvt<<<(NW + 255) / 256, 256, 0, stream>>>(Wo, Wob, NW);

  // 3) projections (WMMA GEMMs); V written transposed for the PV pass
  k_gemm<0><<<dim3(NQ_ / 64, D_ / 64), 128, 0, stream>>>(
      Qln, Wqb, bq, Qp, nullptr, NQ_, D_, D_, D_);
  k_gemm<0><<<dim3(SP_ / 64, D_ / 64), 128, 0, stream>>>(
      Kln, Wkb, bk, Kp, nullptr, SP_, D_, D_, D_);
  k_gemm<1><<<dim3(SP_ / 64, D_ / 64), 128, 0, stream>>>(
      Vln, Wvb, bv, Vt, nullptr, SP_, D_, D_, SP_);

  // 4) flash attention over the bank
  k_attn<<<dim3(NQ_ / 16, H_), 32, 0, stream>>>(Qp, Kp, Vt, bmask, O);

  // 5) mean-pool over TQ, then output projection -> f32 d_out (B,L,D)
  k_pool<<<(NBL_ * D_ + 255) / 256, 256, 0, stream>>>(O, Pl);
  k_gemm<2><<<dim3(NBL_ / 64, D_ / 64), 128, 0, stream>>>(
      Pl, Wob, bo, nullptr, (float*)d_out, NBL_, D_, D_, D_);
}
